// MultiHeadAttention_67542655697646
// MI455X (gfx1250) — matrix operand order repaired and checked by reading, not compiled, not run on hardware
//
#include <hip/hip_runtime.h>
#include <math.h>
#include <stdint.h>

#define BB 8
#define SS 2048
#define HH 256
#define NHEAD 8
#define HD 32

typedef __attribute__((ext_vector_type(16))) _Float16 v16h;
typedef __attribute__((ext_vector_type(8)))  float    v8f;
typedef __attribute__((ext_vector_type(4)))  float    v4f;

union Frag {
  v16h     h;
  uint32_t u[8];
  uint4    q[2];
};

union H2 { uint32_t u; _Float16 h[2]; };

__device__ __forceinline__ uint32_t pkh(float a, float b) {
  H2 x; x.h[0] = (_Float16)a; x.h[1] = (_Float16)b; return x.u;
}

__device__ __forceinline__ v8f wmma16(const Frag& A, const Frag& B, v8f C) {
  // D = A(16x32 f16) * B(32x16 f16) + C(16x16 f32)
  return __builtin_amdgcn_wmma_f32_16x16x32_f16(
      /*neg_a=*/false, A.h, /*neg_b=*/false, B.h,
      /*c_mod=*/(short)0, C, /*reuse_a=*/false, /*reuse_b=*/false);
}

// ---------------------------------------------------------------------------
// Kernel 0: pre-swizzle Wq/Wk/Wv (f32 row-major [256,256], used as X @ W^T)
// into WMMA B-fragment order, f16. Layout: [mat][it(8)][jt(16)][lane(32)][j(8)]
// dwords, so the GEMM loads two contiguous uint4 per fragment per lane.
// ---------------------------------------------------------------------------
__global__ __launch_bounds__(256) void prepack_w(
    const float* __restrict__ Wq, const float* __restrict__ Wk,
    const float* __restrict__ Wv, uint32_t* __restrict__ wfrag) {
  int tid  = blockIdx.x * 256 + threadIdx.x;   // 98304 total
  int mat  = tid >> 15;
  int r    = tid & 32767;
  int tile = r >> 8;                           // it*16 + jt
  int lane = (r >> 3) & 31;
  int j    = r & 7;
  int it   = tile >> 4;
  int jt   = tile & 15;
  const float* W = (mat == 0) ? Wq : ((mat == 1) ? Wk : Wv);
  int row = jt * 16 + (lane & 15);                       // jout (N)
  int col = it * 32 + ((lane < 16) ? 0 : 8) + ((j >> 2) * 16) + 2 * (j & 3);    // i (K)
  wfrag[tid] = pkh(W[row * HH + col], W[row * HH + col + 1]);
}

// ---------------------------------------------------------------------------
// Kernel 1: QKV projection. One wave = 16x64 output tile of X @ W^T + b,
// 32 WMMAs. All 4 B fragments are loaded as one clause before the 4 WMMAs so
// the b128 loads pipeline behind the matrix ops.
// Outputs: qh/kh [bh][s][32] f16 ; vt [bh][32][s] f16 (V transposed).
// ---------------------------------------------------------------------------
__global__ __launch_bounds__(256) void qkv_proj(
    const float* __restrict__ X0, const float* __restrict__ X1,
    const float* __restrict__ X2,
    const float* __restrict__ b0, const float* __restrict__ b1,
    const float* __restrict__ b2,
    const uint32_t* __restrict__ wfrag,
    _Float16* __restrict__ qh, _Float16* __restrict__ kh,
    _Float16* __restrict__ vt) {
  int wid  = (blockIdx.x * 256 + threadIdx.x) >> 5;   // 12288 waves
  int lane = threadIdx.x & 31;
  int mat  = wid >> 12;                                // 0=q 1=k 2=v
  int rem  = wid & 4095;
  int rowt = rem >> 2;                                 // 16-row tile
  int cg   = rem & 3;                                  // 64-col group
  const float* X    = (mat == 0) ? X0 : ((mat == 1) ? X1 : X2);
  const float* bias = (mat == 0) ? b0 : ((mat == 1) ? b1 : b2);
  int s0   = rowt * 16;
  int hsel = lane >> 4;
  int srow = s0 + (lane & 15);
  const float4* Xr = (const float4*)(X + (size_t)srow * HH);

  v8f acc[4] = {};
  for (int it = 0; it < 8; ++it) {
    // issue all loads for this K-chunk first: 4 x float4 (A) + 8 x uint4 (B)
    float4 a0 = Xr[it * 8 + 2 * hsel];
    float4 a1 = Xr[it * 8 + 2 * hsel + 1];
    float4 a2 = Xr[it * 8 + 4 + 2 * hsel];
    float4 a3 = Xr[it * 8 + 4 + 2 * hsel + 1];
    const uint32_t* wb =
        wfrag + (size_t)(mat * 128 + it * 16 + cg * 4) * 256 + lane * 8;
    Frag bf[4];
#pragma unroll
    for (int t = 0; t < 4; ++t) {
      const uint4* bp = (const uint4*)(wb + (size_t)t * 256);
      bf[t].q[0] = bp[0];
      bf[t].q[1] = bp[1];
    }
    Frag af;
    af.u[0] = pkh(a0.x, a0.y); af.u[1] = pkh(a0.z, a0.w);
    af.u[2] = pkh(a1.x, a1.y); af.u[3] = pkh(a1.z, a1.w);
    af.u[4] = pkh(a2.x, a2.y); af.u[5] = pkh(a2.z, a2.w);
    af.u[6] = pkh(a3.x, a3.y); af.u[7] = pkh(a3.z, a3.w);
#pragma unroll
    for (int t = 0; t < 4; ++t) acc[t] = wmma16(af, bf[t], acc[t]);
  }

  int b  = s0 >> 11;          // batch; 16-row tiles never span batches
  int sl = s0 & (SS - 1);
  for (int t = 0; t < 4; ++t) {
    int jcol = cg * 64 + t * 16 + (lane & 15);   // output feature (C-layout N)
    float bj = bias[jcol];
    int h = jcol >> 5, d = jcol & 31;
    if (mat < 2) {
      _Float16* out =
          ((mat == 0) ? qh : kh) + (size_t)(b * NHEAD + h) * SS * HD;
#pragma unroll
      for (int vi = 0; vi < 8; ++vi) {
        int s = sl + vi + hsel * 8;              // C-layout M
        out[(size_t)s * HD + d] = (_Float16)(acc[t][vi] + bj);
      }
    } else {
      // vt[bh][d][s]: 8 consecutive s per lane -> one b128 store
      size_t base = ((size_t)(b * NHEAD + h) * HD + d) * SS + sl + hsel * 8;
      uint4 pk;
      pk.x = pkh(acc[t][0] + bj, acc[t][1] + bj);
      pk.y = pkh(acc[t][2] + bj, acc[t][3] + bj);
      pk.z = pkh(acc[t][4] + bj, acc[t][5] + bj);
      pk.w = pkh(acc[t][6] + bj, acc[t][7] + bj);
      *(uint4*)(vt + base) = pk;
    }
  }
}

// ---------------------------------------------------------------------------
// Kernel 2: attention. Block = 2 waves; each wave owns 16 query rows of one
// (b,h). Score row-block lives in LDS as f16 with padded row stride
// (1028 dwords). Dynamic LDS: 131840 B. Weight/context stores use NT hints
// so the 1 GB write-once stream does not evict the L2-resident K/V.
// ---------------------------------------------------------------------------
__global__ __launch_bounds__(64) void attn(
    const _Float16* __restrict__ qh, const _Float16* __restrict__ kh,
    const _Float16* __restrict__ vt, const unsigned char* __restrict__ mask,
    float* __restrict__ weight, float* __restrict__ context) {
  extern __shared__ uint32_t smem[];
  int lane = threadIdx.x & 31;
  int wave = threadIdx.x >> 5;
  int bh   = blockIdx.x >> 6;
  int qt2  = blockIdx.x & 63;
  int b    = bh >> 3;
  int h    = bh & 7;
  int q0   = qt2 * 32 + wave * 16;
  uint32_t* sc   = smem + wave * 16448;            // 16 rows x 1028 dwords
  _Float16* sch  = (_Float16*)sc;                  // row stride 2056 halves
  float*    rmax = (float*)(smem + 32896) + wave * 16;
  float*    rsum = (float*)(smem + 32928) + wave * 16;
  int hsel = lane >> 4;
  int nlo  = lane & 15;

  const _Float16* qhp = qh + (size_t)bh * SS * HD;
  const _Float16* khp = kh + (size_t)bh * SS * HD;
  const _Float16* vtp = vt + (size_t)bh * HD * SS;
  const unsigned char* mrow = mask + (size_t)b * SS * SS;
  float* wout = weight + ((size_t)bh * SS + q0) * SS;

  // Q fragment (A layout): lane<16 row q0+nlo cols 0..15 (lane>=16: 16..31)
  Frag qf;
  {
    const uint4* qr = (const uint4*)(qhp + (size_t)(q0 + nlo) * HD);
    qf.q[0] = qr[2 * hsel];
    qf.q[1] = qr[2 * hsel + 1];
  }
  const float scale = 0.17677669529663687f;  // 1/sqrt(HD)

  // ---- scores: 16x16 tile = Q(16x32) x K^T(32x16), K frag double-buffered --
  Frag kf;
  {
    const uint4* kr = (const uint4*)(khp + (size_t)nlo * HD);
    kf.q[0] = kr[2 * hsel];
    kf.q[1] = kr[2 * hsel + 1];
  }
  for (int ct = 0; ct < 128; ++ct) {
    int ncol = ct * 16 + nlo;
    // prefetch mask bytes for this tile (independent of the WMMA)
    unsigned char mk[8];
#pragma unroll
    for (int vi = 0; vi < 8; ++vi)
      mk[vi] = mrow[(size_t)(q0 + vi + hsel * 8) * SS + ncol];
    v8f c = {};
    c = wmma16(qf, kf, c);
    // prefetch next K fragment while the WMMA result is still in flight
    if (ct + 1 < 128) {
      const uint4* kr =
          (const uint4*)(khp + (size_t)((ct + 1) * 16 + nlo) * HD);
      kf.q[0] = kr[2 * hsel];
      kf.q[1] = kr[2 * hsel + 1];
    }
#pragma unroll
    for (int vi = 0; vi < 8; ++vi) {
      int r = vi + hsel * 8;
      float sv = c[vi] * scale;
      sch[r * 2056 + ncol] =
          mk[vi] ? (_Float16)sv : (_Float16)(-__builtin_huge_valf());
    }
  }

  // ---- row max (2 dwords = 4 halves per lane per step) ----
  for (int r = 0; r < 16; ++r) {
    float m = -3.0e38f;
    const uint32_t* sr = sc + r * 1028;
    for (int cc = lane * 2; cc < 1024; cc += 64) {
      H2 x0, x1; x0.u = sr[cc]; x1.u = sr[cc + 1];
      m = fmaxf(m, fmaxf((float)x0.h[0], (float)x0.h[1]));
      m = fmaxf(m, fmaxf((float)x1.h[0], (float)x1.h[1]));
    }
    for (int off = 16; off; off >>= 1) m = fmaxf(m, __shfl_xor(m, off, 32));
    if (lane == 0) rmax[r] = m;
  }

  // ---- exp + row sum (store e back to LDS) ----
  for (int r = 0; r < 16; ++r) {
    float m = rmax[r];
    uint32_t* sr = sc + r * 1028;
    float sum = 0.0f;
    for (int cc = lane * 2; cc < 1024; cc += 64) {
      H2 x0, x1; x0.u = sr[cc]; x1.u = sr[cc + 1];
      float e0 = __expf((float)x0.h[0] - m);
      float e1 = __expf((float)x0.h[1] - m);
      float e2 = __expf((float)x1.h[0] - m);
      float e3 = __expf((float)x1.h[1] - m);
      sum += (e0 + e1) + (e2 + e3);
      sr[cc]     = pkh(e0, e1);
      sr[cc + 1] = pkh(e2, e3);
    }
    for (int off = 16; off; off >>= 1) sum += __shfl_xor(sum, off, 32);
    if (lane == 0) rsum[r] = sum;
  }

  // ---- normalize: b128 NT weight stores + f16 p back to LDS ----
  for (int r = 0; r < 16; ++r) {
    float inv = 1.0f / rsum[r];
    uint32_t* sr = sc + r * 1028;
    float* wr = wout + (size_t)r * SS;
    for (int cc = lane * 2; cc < 1024; cc += 64) {
      H2 x0, x1; x0.u = sr[cc]; x1.u = sr[cc + 1];
      v4f pp;
      pp.x = (float)x0.h[0] * inv;
      pp.y = (float)x0.h[1] * inv;
      pp.z = (float)x1.h[0] * inv;
      pp.w = (float)x1.h[1] * inv;
      __builtin_nontemporal_store(pp, (v4f*)(wr + 2 * cc));
      sr[cc]     = pkh(pp.x, pp.y);
      sr[cc + 1] = pkh(pp.z, pp.w);
    }
  }

  // ---- context = P(16x2048) @ V(2048x32), 2 WMMAs per 32-wide K chunk ----
  v8f a0 = {}, a1 = {};
  for (int kt = 0; kt < 64; ++kt) {
    // issue all loads first (2 LDS b128 + 4 global b128), then 2 WMMAs
    Frag pf;  // A fragment from LDS p
    const uint32_t* pb = sc + nlo * 1028 + kt * 16 + hsel * 8;
    pf.q[0] = *(const uint4*)pb;
    pf.q[1] = *(const uint4*)(pb + 4);
    Frag vf0, vf1;  // B fragments from vt[bh][d][s]
    const uint4* v0 = (const uint4*)(vtp + (size_t)nlo * SS) + kt * 4 + hsel * 2;
    const uint4* v1 =
        (const uint4*)(vtp + (size_t)(16 + nlo) * SS) + kt * 4 + hsel * 2;
    vf0.q[0] = v0[0]; vf0.q[1] = v0[1];
    vf1.q[0] = v1[0]; vf1.q[1] = v1[1];
    a0 = wmma16(pf, vf0, a0);
    a1 = wmma16(pf, vf1, a1);
  }

  // ---- store context [b, s, h*32 + n] (NT: written once) ----
  float* ctx = context + ((size_t)b * SS + q0) * HH + h * HD;
#pragma unroll
  for (int vi = 0; vi < 8; ++vi) {
    int r = vi + hsel * 8;
    float* row = ctx + (size_t)r * HH;
    __builtin_nontemporal_store(a0[vi], row + nlo);
    __builtin_nontemporal_store(a1[vi], row + 16 + nlo);
  }
}

// ---------------------------------------------------------------------------
extern "C" void kernel_launch(void* const* d_in, const int* in_sizes, int n_in,
                              void* d_out, int out_size, void* d_ws,
                              size_t ws_size, hipStream_t stream) {
  (void)in_sizes; (void)n_in; (void)out_size; (void)ws_size;
  const float* q  = (const float*)d_in[0];
  const float* k  = (const float*)d_in[1];
  const float* v  = (const float*)d_in[2];
  const unsigned char* mask = (const unsigned char*)d_in[3];
  const float* Wq = (const float*)d_in[4];
  const float* bq = (const float*)d_in[5];
  const float* Wk = (const float*)d_in[6];
  const float* bk = (const float*)d_in[7];
  const float* Wv = (const float*)d_in[8];
  const float* bv = (const float*)d_in[9];

  float* context = (float*)d_out;                         // [8,2048,256]
  float* weight  = context + (size_t)BB * SS * HH;        // [8,8,2048,2048]

  uint8_t* ws = (uint8_t*)d_ws;
  uint32_t* wfrag = (uint32_t*)ws;                        // 384 KB
  _Float16* qh = (_Float16*)(ws + (512u << 10));          // 8 MB
  _Float16* kh = qh + (size_t)BB * NHEAD * SS * HD;       // 8 MB
  _Float16* vt = kh + (size_t)BB * NHEAD * SS * HD;       // 8 MB

  prepack_w<<<384, 256, 0, stream>>>(Wq, Wk, Wv, wfrag);
  qkv_proj<<<1536, 256, 0, stream>>>(q, k, v, bq, bk, bv, wfrag, qh, kh, vt);
  attn<<<4096, 64, 131840, stream>>>(qh, kh, vt, mask, weight, context);
}
